// GATClassifier_80633716015355
// MI455X (gfx1250) — matrix operand order repaired and checked by reading, not compiled, not run on hardware
//
#include <hip/hip_runtime.h>
#include <hip/hip_bf16.h>

#define N_CLASSES 4096
#define HIDDEN    512
#define N_HEADS   4
#define HEAD_DIM  128
#define TEXT_DIM  768
#define BATCH     32
#define ALPHA_LK  0.2f
#define TILE_ROWS 128   // rows of output per attention workgroup (8 waves * 16)

typedef __attribute__((ext_vector_type(16))) __bf16         v16bf;
typedef __attribute__((ext_vector_type(16))) unsigned short v16u;
typedef __attribute__((ext_vector_type(8)))  float          v8f;

// TDM descriptor group types (probe-verified signatures)
typedef __attribute__((ext_vector_type(4))) unsigned int tdm_v4u;
typedef __attribute__((ext_vector_type(8))) int          tdm_v8i;
typedef __attribute__((ext_vector_type(4))) int          tdm_v4i;

#if defined(__has_builtin)
#if __has_builtin(__builtin_amdgcn_tensor_load_to_lds) && __has_builtin(__builtin_amdgcn_s_wait_tensorcnt)
#define HAVE_TDM 1
#endif
#endif
#ifndef HAVE_TDM
#define HAVE_TDM 0
#endif

union U16x16 { v16u u; v16bf b; };
union F4 { float4 v; float s[4]; };
union I4 { int4 v; int s[4]; };

__device__ __forceinline__ unsigned short f2bf_u(float f) {
  union { float f; unsigned u; } v; v.f = f;
  unsigned r = v.u + 0x7FFFu + ((v.u >> 16) & 1u);   // round-to-nearest-even
  return (unsigned short)(r >> 16);
}
__device__ __forceinline__ float bf2f(unsigned short s) {
  union { unsigned u; float f; } v; v.u = ((unsigned)s) << 16; return v.f;
}
__device__ __forceinline__ v8f v8f_zero() {
  v8f z;
#pragma unroll
  for (int e = 0; e < 8; ++e) z[e] = 0.0f;
  return z;
}

// A fragment (16x32 bf16, M x K) from an f32 row-major source.
// p = row base + k0 + 16*hi. Elements 0..15 <- p[0..15].
__device__ __forceinline__ v16bf a_frag_from_f32(const float* __restrict__ p) {
  F4 c0, c1, c2, c3;
  c0.v = *reinterpret_cast<const float4*>(p);
  c1.v = *reinterpret_cast<const float4*>(p + 4);
  c2.v = *reinterpret_cast<const float4*>(p + 8);
  c3.v = *reinterpret_cast<const float4*>(p + 12);
  v16u u;
#pragma unroll
  for (int e = 0; e < 4; ++e) {
    u[e]      = f2bf_u(c0.s[e]);
    u[4 + e]  = f2bf_u(c1.s[e]);
    u[8 + e]  = f2bf_u(c2.s[e]);
    u[12 + e] = f2bf_u(c3.s[e]);
  }
  U16x16 c; c.u = u; return c.b;
}

// B fragment (32x16 bf16, K x N): 16 contiguous K-elements per lane at
// p = col_base + k0 + 16*hi (32B aligned).
__device__ __forceinline__ v16bf b_frag_from_bf16(const unsigned short* p) {
  U16x16 c;
  c.u = *reinterpret_cast<const v16u*>(p);
  return c.b;
}

// ---------------- prepack kernels (once) ----------------
__global__ void pack_wtc_kernel(const float* __restrict__ Wtc, unsigned short* __restrict__ WT) {
  int idx = blockIdx.x * blockDim.x + threadIdx.x;      // over 4096*768
  int n = idx / TEXT_DIM, k = idx % TEXT_DIM;
  WT[idx] = f2bf_u(Wtc[(size_t)k * N_CLASSES + n]);     // WT[n][k]
}
__global__ void pack_ws_kernel(const float* __restrict__ Ws, unsigned short* __restrict__ WT) {
  int idx = blockIdx.x * blockDim.x + threadIdx.x;      // over 2*4*128*512
  int d = idx % HIDDEN;
  int rest = idx / HIDDEN;
  int f = rest % HEAD_DIM;
  int lh = rest / HEAD_DIM;                             // l*4 + h
  WT[idx] = f2bf_u(Ws[((size_t)lh * HIDDEN + d) * HEAD_DIM + f]);  // WT[lh][f][d]
}

// ---------------- text logits GEMM (32x768 @ 768x4096), WMMA bf16 ----------------
__global__ void textgemm_kernel(const float* __restrict__ T,
                                const unsigned short* __restrict__ WtcT,
                                float* __restrict__ tl) {
  const int lane = threadIdx.x, m = lane & 15, hi = lane >> 4;
  const int n0 = blockIdx.x * 16;
  const int m0 = blockIdx.y * 16;
  const float* trow = T + (size_t)(m0 + m) * TEXT_DIM;
  const unsigned short* wrow = WtcT + (size_t)(n0 + m) * TEXT_DIM;
  v8f acc = v8f_zero();
  for (int k0 = 0; k0 < TEXT_DIM; k0 += 32) {
    v16bf a = a_frag_from_f32(trow + k0 + 16 * hi);
    v16bf b = b_frag_from_bf16(wrow + k0 + 16 * hi);
    acc = __builtin_amdgcn_wmma_f32_16x16x32_bf16(false, a, false, b, (short)0, acc, false, false);
  }
#pragma unroll
  for (int r = 0; r < 8; ++r)
    tl[(size_t)(m0 + r + 8 * hi) * N_CLASSES + n0 + m] = acc[r];
}

// ---------------- per-head projection: hT[h][f][n] = bf16( (X @ W_h)[n][f] ) ----------------
__global__ void proj_kernel(const float* __restrict__ X,
                            const unsigned short* __restrict__ WsT_layer, // [H][128][512]
                            unsigned short* __restrict__ hT) {            // [H][128][4096]
  const int lane = threadIdx.x, m = lane & 15, hi = lane >> 4;
  const int n0 = blockIdx.x * 16;   // node tile
  const int f0 = blockIdx.y * 16;   // feature tile
  const int h  = blockIdx.z;
  const float* xrow = X + (size_t)(n0 + m) * HIDDEN;
  const unsigned short* wcol = WsT_layer + ((size_t)h * HEAD_DIM + f0 + m) * HIDDEN;
  v8f acc = v8f_zero();
#pragma unroll 4
  for (int k0 = 0; k0 < HIDDEN; k0 += 32) {
    v16bf a = a_frag_from_f32(xrow + k0 + 16 * hi);
    v16bf b = b_frag_from_bf16(wcol + k0 + 16 * hi);
    acc = __builtin_amdgcn_wmma_f32_16x16x32_bf16(false, a, false, b, (short)0, acc, false, false);
  }
  unsigned short* orow = hT + ((size_t)h * HEAD_DIM + f0 + m) * N_CLASSES + n0 + 8 * hi;
#pragma unroll
  for (int r = 0; r < 8; ++r) orow[r] = f2bf_u(acc[r]);
}

// ---------------- f1/f2 = h . a1 / h . a2 ----------------
__global__ void f12_kernel(const unsigned short* __restrict__ hT,
                           const float* __restrict__ A,   // As + l*H*256 : [H][2*128]
                           float* __restrict__ f1, float* __restrict__ f2) {
  int t = blockIdx.x * blockDim.x + threadIdx.x;          // H * N
  int h = t >> 12, n = t & (N_CLASSES - 1);
  const float* a1 = A + (size_t)h * 2 * HEAD_DIM;
  const float* a2 = a1 + HEAD_DIM;
  const unsigned short* hp = hT + (size_t)h * HEAD_DIM * N_CLASSES + n;
  float s1 = 0.0f, s2 = 0.0f;
#pragma unroll 4
  for (int f = 0; f < HEAD_DIM; ++f) {
    float v = bf2f(hp[(size_t)f * N_CLASSES]);
    s1 += v * a1[f];
    s2 += v * a2[f];
  }
  f1[t] = s1; f2[t] = s2;
}

// ---------------- per-row masked softmax stats: m_i, 1/l_i ----------------
__global__ void stats_kernel(const int* __restrict__ adj,
                             const float* __restrict__ f1, const float* __restrict__ f2,
                             float* __restrict__ mbuf, float* __restrict__ linv) {
  const int i = blockIdx.x, h = blockIdx.y, lane = threadIdx.x;
  const float f1i = f1[h * N_CLASSES + i];
  const float* f2h = f2 + (size_t)h * N_CLASSES;
  const int* arow = adj + (size_t)i * N_CLASSES;
  float m = -3.0e38f, l = 0.0f;
  for (int j = lane; j < N_CLASSES; j += 32) {
    if (arow[j] != 0) {
      float x = f1i + f2h[j];
      x = x > 0.0f ? x : ALPHA_LK * x;
      if (x > m) { l = l * __expf(m - x) + 1.0f; m = x; }
      else        l += __expf(x - m);
    }
  }
#pragma unroll
  for (int off = 16; off > 0; off >>= 1) {
    float mo = __shfl_xor(m, off, 32);
    float lo = __shfl_xor(l, off, 32);
    float mn = fmaxf(m, mo);
    l = l * __expf(m - mn) + lo * __expf(mo - mn);
    m = mn;
  }
  if (lane == 0) {
    mbuf[h * N_CLASSES + i] = m;
    linv[h * N_CLASSES + i] = 1.0f / l;
  }
}

// ---------------- attention: 8 waves/block share the 32x128 bf16 V tile in LDS ----------------
// TDM (tensor_load_to_lds) double-buffers the tile; each wave does 8 WMMA per j-step.
__global__ void __launch_bounds__(256)
attn_kernel(const int* __restrict__ adj,
            const unsigned short* __restrict__ hT,  // [H][128][4096] bf16
            const float* __restrict__ f1, const float* __restrict__ f2,
            const float* __restrict__ mbuf, const float* __restrict__ linv,
            float* __restrict__ xout) {             // [4096][512]
  __shared__ __align__(32) unsigned short smem[2][HEAD_DIM * 32];  // 2 x 8KB

  const int tid = threadIdx.x;
  const int wv = tid >> 5;
  const int lane = tid & 31, m = lane & 15, hi = lane >> 4;
  const int h  = blockIdx.y;
  const int i0 = blockIdx.x * TILE_ROWS + wv * 16;
  const int row = i0 + m;
  const float f1r = f1[h * N_CLASSES + row];
  const float mr  = mbuf[h * N_CLASSES + row];
  const float il  = linv[h * N_CLASSES + row];
  const int* arow = adj + (size_t)row * N_CLASSES;
  const float* f2h = f2 + (size_t)h * N_CLASSES;
  const unsigned short* hTh = hT + (size_t)h * HEAD_DIM * N_CLASSES;

  v8f acc[8];
#pragma unroll
  for (int t = 0; t < 8; ++t) acc[t] = v8f_zero();

#if HAVE_TDM
  // --- TDM descriptor issue: 2D tile, tile_dim0=32 (j, contiguous), tile_dim1=128 (f rows),
  //     data_size=2B, tensor_dim0_stride=4096 elements, LDS written densely as [f][32]. ---
  const unsigned ldsA[2] = { (unsigned)(size_t)&smem[0][0], (unsigned)(size_t)&smem[1][0] };
  auto tdm_issue = [&](int buf, int j0) {
    unsigned long long ga = (unsigned long long)(size_t)(hTh + j0);
    tdm_v4u g0;
    g0[0] = 1u;                                            // count=1, user descriptor
    g0[1] = ldsA[buf];                                     // lds_addr
    g0[2] = (unsigned)(ga & 0xffffffffu);                  // global_addr[31:0]
    g0[3] = (unsigned)((ga >> 32) & 0x01ffffffu) | 0x80000000u;  // addr[56:32] | type=2
    tdm_v8i g1;
    g1[0] = (int)(1u << 16);                               // data_size=1 (2 bytes)
    g1[1] = (int)(((unsigned)N_CLASSES & 0xffffu) << 16);  // tensor_dim0[15:0]
    g1[2] = (int)(((unsigned)N_CLASSES >> 16) | (((unsigned)HEAD_DIM & 0xffffu) << 16)); // dim0 hi | dim1 lo
    g1[3] = (int)(32u << 16);                              // tensor_dim1 hi | tile_dim0=32
    g1[4] = (int)(unsigned)HEAD_DIM;                       // tile_dim1=128, tile_dim2=0
    g1[5] = (int)(unsigned)N_CLASSES;                      // tensor_dim0_stride[31:0]
    g1[6] = 0;                                             // stride hi | dim1_stride (unused, 2D)
    g1[7] = 0;
    tdm_v4i gz = {0, 0, 0, 0};
#if __clang_major__ >= 23
    tdm_v8i gz8 = {0, 0, 0, 0, 0, 0, 0, 0};
    __builtin_amdgcn_tensor_load_to_lds(g0, g1, gz, gz, gz8, 0);
#else
    __builtin_amdgcn_tensor_load_to_lds(g0, g1, gz, gz, 0);
#endif
  };
  if (wv == 0) tdm_issue(0, 0);
  int cur = 0;
#endif

  for (int j0 = 0; j0 < N_CLASSES; j0 += 32) {
#if HAVE_TDM
    if (wv == 0) {
      if (j0 + 32 < N_CLASSES) {
        tdm_issue(cur ^ 1, j0 + 32);                 // prefetch next tile
        __builtin_amdgcn_s_wait_tensorcnt(1);        // current tile complete
      } else {
        __builtin_amdgcn_s_wait_tensorcnt(0);
      }
    }
    __syncthreads();                                 // LDS tile visible to all waves
    const unsigned short* sb = smem[cur];
#else
    __syncthreads();                                 // previous iter's readers done
    {
      const int f = tid >> 1, half = tid & 1;        // 256 threads move 8KB
      const uint4* src = reinterpret_cast<const uint4*>(hTh + (size_t)f * N_CLASSES + j0 + half * 16);
      uint4* dst = reinterpret_cast<uint4*>(&smem[0][f * 32 + half * 16]);
      dst[0] = src[0];
      dst[1] = src[1];
    }
    __syncthreads();
    const unsigned short* sb = smem[0];
#endif

    __builtin_prefetch(arow + j0 + 64, 0, 0);        // global_prefetch_b8 on the adj stream
    const int jb = j0 + 16 * hi;
    I4 ad[4]; F4 ff[4];
    ad[0].v = *reinterpret_cast<const int4*>(arow + jb);
    ad[1].v = *reinterpret_cast<const int4*>(arow + jb + 4);
    ad[2].v = *reinterpret_cast<const int4*>(arow + jb + 8);
    ad[3].v = *reinterpret_cast<const int4*>(arow + jb + 12);
    ff[0].v = *reinterpret_cast<const float4*>(f2h + jb);
    ff[1].v = *reinterpret_cast<const float4*>(f2h + jb + 4);
    ff[2].v = *reinterpret_cast<const float4*>(f2h + jb + 8);
    ff[3].v = *reinterpret_cast<const float4*>(f2h + jb + 12);
    v16u pu;
#pragma unroll
    for (int e = 0; e < 16; ++e) {
      const int c = e >> 2, s = e & 3;
      float pp = 0.0f;
      if (ad[c].s[s] != 0) {
        float x = f1r + ff[c].s[s];
        x = x > 0.0f ? x : ALPHA_LK * x;
        pp = __expf(x - mr) * il;                    // already normalized probability
      }
      pu[e] = f2bf_u(pp);
    }
    U16x16 pc; pc.u = pu;
    const v16bf a = pc.b;
    const unsigned short* bbase = sb + 16 * hi;      // [f][32] layout, ld = 32
#pragma unroll
    for (int ft = 0; ft < 8; ++ft) {
      v16bf b = b_frag_from_bf16(bbase + (size_t)(ft * 16 + m) * 32);
      acc[ft] = __builtin_amdgcn_wmma_f32_16x16x32_bf16(false, a, false, b, (short)0, acc[ft], false, false);
    }

#if HAVE_TDM
    __syncthreads();                                 // all reads done before TDM reuses buffer
    cur ^= 1;
#endif
  }

#pragma unroll
  for (int ft = 0; ft < 8; ++ft) {
#pragma unroll
    for (int r = 0; r < 8; ++r) {
      float v = acc[ft][r];
      v = v > 0.0f ? v : (__expf(v) - 1.0f);         // elu
      const int node = i0 + r + 8 * hi;
      xout[(size_t)node * HIDDEN + h * HEAD_DIM + ft * 16 + m] = v;
    }
  }
}

// ---------------- final: out[b][n] = tl[b][n] + b_tc[n] + (x[n].W_out + b_out) ----------------
__global__ void final_kernel(const float* __restrict__ x, const float* __restrict__ Wout,
                             const float* __restrict__ bout, const float* __restrict__ tl,
                             const float* __restrict__ btc, float* __restrict__ out) {
  const int n = blockIdx.x, lane = threadIdx.x;
  const float* xr = x + (size_t)n * HIDDEN;
  float s = 0.0f;
  for (int d = lane; d < HIDDEN; d += 32) s += xr[d] * Wout[d];
#pragma unroll
  for (int off = 16; off > 0; off >>= 1) s += __shfl_xor(s, off, 32);
  const float cs = s + bout[0] + btc[n];
  out[(size_t)lane * N_CLASSES + n] = tl[(size_t)lane * N_CLASSES + n] + cs;
}

extern "C" void kernel_launch(void* const* d_in, const int* in_sizes, int n_in,
                              void* d_out, int out_size, void* d_ws, size_t ws_size,
                              hipStream_t stream) {
  (void)in_sizes; (void)n_in; (void)out_size; (void)ws_size;
  const float* text = (const float*)d_in[0];
  const int*   adj  = (const int*)d_in[1];
  const float* Wtc  = (const float*)d_in[2];
  const float* btc  = (const float*)d_in[3];
  const float* cemb = (const float*)d_in[4];
  const float* Ws   = (const float*)d_in[5];
  const float* As   = (const float*)d_in[6];
  const float* Wout = (const float*)d_in[7];
  const float* bout = (const float*)d_in[8];
  float* out = (float*)d_out;

  char* p = (char*)d_ws;
  auto alloc = [&](size_t bytes) -> char* {
    char* r = p; p += (bytes + 255) & ~(size_t)255; return r;
  };
  float*          tl   = (float*)alloc((size_t)BATCH * N_CLASSES * 4);
  unsigned short* WtcT = (unsigned short*)alloc((size_t)N_CLASSES * TEXT_DIM * 2);
  unsigned short* WsT  = (unsigned short*)alloc((size_t)2 * N_HEADS * HEAD_DIM * HIDDEN * 2);
  unsigned short* hT   = (unsigned short*)alloc((size_t)N_HEADS * HEAD_DIM * N_CLASSES * 2);
  float*          f1   = (float*)alloc((size_t)N_HEADS * N_CLASSES * 4);
  float*          f2   = (float*)alloc((size_t)N_HEADS * N_CLASSES * 4);
  float*          mbuf = (float*)alloc((size_t)N_HEADS * N_CLASSES * 4);
  float*          linv = (float*)alloc((size_t)N_HEADS * N_CLASSES * 4);
  float*          x1   = (float*)alloc((size_t)N_CLASSES * HIDDEN * 4);
  float*          x2   = (float*)alloc((size_t)N_CLASSES * HIDDEN * 4);

  pack_wtc_kernel<<<(N_CLASSES * TEXT_DIM) / 256, 256, 0, stream>>>(Wtc, WtcT);
  pack_ws_kernel<<<(2 * N_HEADS * HEAD_DIM * HIDDEN) / 256, 256, 0, stream>>>(Ws, WsT);
  textgemm_kernel<<<dim3(N_CLASSES / 16, BATCH / 16), 32, 0, stream>>>(text, WtcT, tl);

  for (int l = 0; l < 2; ++l) {
    const float* X = (l == 0) ? cemb : x1;
    float* Xo = (l == 0) ? x1 : x2;
    const unsigned short* WsT_l = WsT + (size_t)l * N_HEADS * HEAD_DIM * HIDDEN;
    const float* As_l = As + (size_t)l * N_HEADS * 2 * HEAD_DIM;
    proj_kernel<<<dim3(N_CLASSES / 16, HEAD_DIM / 16, N_HEADS), 32, 0, stream>>>(X, WsT_l, hT);
    f12_kernel<<<(N_HEADS * N_CLASSES) / 256, 256, 0, stream>>>(hT, As_l, f1, f2);
    stats_kernel<<<dim3(N_CLASSES, N_HEADS), 32, 0, stream>>>(adj, f1, f2, mbuf, linv);
    attn_kernel<<<dim3(N_CLASSES / TILE_ROWS, N_HEADS), 256, 0, stream>>>(adj, hT, f1, f2, mbuf, linv, Xo);
  }

  final_kernel<<<N_CLASSES, 32, 0, stream>>>(x2, Wout, bout, tl, btc, out);
}
